// TAGM_GRU_15247133901426
// MI455X (gfx1250) — matrix operand order repaired and checked by reading, not compiled, not run on hardware
//
#include <hip/hip_runtime.h>
#include <hip/hip_bf16.h>
#include <math.h>

// Problem dims
static constexpr int TT = 512;
static constexpr int BB = 64;
static constexpr int II = 512;
static constexpr int HH = 1024;
static constexpr int CC = 128;
static constexpr int KT_H  = 32;   // k-tiles covering h (1024 / 32)
static constexpr int KT_ALL = 48;  // k-tiles covering [h|x] (1536 / 32)
static constexpr float SIG_W = 3.0f;

typedef __bf16 bf16_t;
typedef bf16_t  bf16x16 __attribute__((ext_vector_type(16)));
typedef float   f32x8   __attribute__((ext_vector_type(8)));
typedef unsigned u32x4  __attribute__((ext_vector_type(4)));

union Frag {
    u32x4   q[2];
    bf16x16 v;
};

__device__ __forceinline__ unsigned short f2bf(float f) {
    unsigned x = __float_as_uint(f);
    unsigned r = x + 0x7FFFu + ((x >> 16) & 1u);   // round-to-nearest-even
    return (unsigned short)(r >> 16);
}
__device__ __forceinline__ float bf2f(unsigned short h) {
    return __uint_as_float(((unsigned)h) << 16);
}
__device__ __forceinline__ float sigm(float x) { return 1.0f / (1.0f + __expf(-x)); }

__device__ __forceinline__ f32x8 zero8() {
    f32x8 z = {0.f,0.f,0.f,0.f,0.f,0.f,0.f,0.f};
    return z;
}

__device__ __forceinline__ f32x8 wmma_bf16(const Frag& a, const Frag& b, f32x8 c) {
    // D = A(16x32 bf16) * B(32x16 bf16) + C(16x16 f32)
    return __builtin_amdgcn_wmma_f32_16x16x32_bf16(
        false, a.v, false, b.v, (short)0, c, false, false);
}

// ---------------------------------------------------------------------------
// x [B,T,I] f32  ->  xbf [T,B,I] bf16  (t-major so each step's tile is dense)
// ---------------------------------------------------------------------------
__global__ void pack_x_kernel(const float* __restrict__ x,
                              unsigned short* __restrict__ xbf) {
    long idx = (long)blockIdx.x * blockDim.x + threadIdx.x;
    if (idx >= (long)TT * BB * II) return;
    int i = (int)(idx % II);
    long r = idx / II;
    int b = (int)(r % BB);
    int t = (int)(r / BB);
    xbf[idx] = f2bf(x[((long)b * TT + t) * II + i]);
}

// ---------------------------------------------------------------------------
// Pack W = [Wh (N x 1024) | Wi (N x 512)] into WMMA B-fragment order:
// dword index = ((nt*48 + kt)*32 + lane)*8 + v
// lane<16 : (k = kt*32 + (v<4 ? 2v : 16 + 2(v-4)),  n = nt*16 + lane)
// lane>=16: (k = kt*32 + 8 + (v<4 ? 2v : 16 + 2(v-4)), n = nt*16 + lane-16)   (per ISA B layout)
// ---------------------------------------------------------------------------
__global__ void pack_w_kernel(const float* __restrict__ wh,
                              const float* __restrict__ wi,
                              unsigned* __restrict__ dst, int ntiles) {
    int idx = blockIdx.x * blockDim.x + threadIdx.x;
    int tile = idx >> 8;
    if (tile >= ntiles * KT_ALL) return;
    int inner = idx & 255;
    int kt = tile % KT_ALL;
    int nt = tile / KT_ALL;
    int lane = inner >> 3;
    int v    = inner & 7;
    int n  = nt * 16 + (lane & 15);
    int k0 = kt * 32 + ((lane & 16) ? 8 : 0) + ((v & 4) ? 16 : 0) + 2 * (v & 3);
    float w0, w1;
    if (k0 < HH) {
        w0 = wh[(long)n * HH + k0];
        w1 = wh[(long)n * HH + k0 + 1];
    } else {
        int kx = k0 - HH;
        w0 = wi[(long)n * II + kx];
        w1 = wi[(long)n * II + kx + 1];
    }
    dst[idx] = (unsigned)f2bf(w0) | ((unsigned)f2bf(w1) << 16);
}

__global__ void zero_kernel(float* __restrict__ p, int n) {
    int idx = blockIdx.x * blockDim.x + threadIdx.x;
    if (idx < n) p[idx] = 0.0f;
}

// ---------------------------------------------------------------------------
// Attention bidirectional GRU. grid.x = 2 (direction), 1024 threads (32 waves).
// h kept in LDS (bf16, ping-pong 2 x 64 x 1024 = 256 KB). Per step, wave w owns
// output columns j in [32w, 32w+32): 2 j-tiles x 4 m-tiles x 3 gates of
// 16x16x32 WMMA over K = 1536 ([h|x]).  Fused logit = h_new . att_fc_w.
// ---------------------------------------------------------------------------
__global__ __launch_bounds__(1024) void att_gru_kernel(
    const unsigned* __restrict__ wf_pack, const unsigned* __restrict__ wb_pack,
    const float* __restrict__ bi_f, const float* __restrict__ bh_f,
    const float* __restrict__ bi_b, const float* __restrict__ bh_b,
    const unsigned short* __restrict__ xbf, const float* __restrict__ fc_w,
    float* __restrict__ logit) {
    extern __shared__ unsigned smem[];   // 2 buffers of 64*512 dwords (bf16 h)
    const int dir = blockIdx.x;
    const unsigned* Wp = dir ? wb_pack : wf_pack;
    const float* bi = dir ? bi_b : bi_f;
    const float* bh = dir ? bh_b : bh_f;

    const int tid  = threadIdx.x;
    const int wave = tid >> 5;
    const int lane = tid & 31;
    const int lh   = lane & 15;
    const int sel4 = (lane & 16) ? 4 : 0;
    const int rowoff = (lane & 16) ? 8 : 0;
    const int HBUF = BB * (HH / 2);      // dwords per h buffer

    for (int i = tid; i < HBUF; i += 1024) smem[i] = 0u;   // h0 = 0
    __syncthreads();

    for (int step = 0; step < TT; ++step) {
        const int t = dir ? (TT - 1 - step) : step;
        const unsigned* hcur = smem + (step & 1) * HBUF;
        unsigned*       hnxt = smem + ((step & 1) ^ 1) * HBUF;
        const unsigned short* hcur16 = (const unsigned short*)hcur;
        unsigned short*       hnxt16 = (unsigned short*)hnxt;

        for (int jt2 = 0; jt2 < 2; ++jt2) {
            const int jt = 2 * wave + jt2;
            for (int mt = 0; mt < 4; ++mt) {
                const int m = mt * 16 + lh;
                f32x8 accR = zero8(), accZ = zero8();
                f32x8 accNH = zero8(), accNX = zero8();

                const unsigned* arow = hcur + m * (HH / 2);
                const unsigned* brR = Wp + (size_t)(jt)       * KT_ALL * 256 + lane * 8;
                const unsigned* brZ = Wp + (size_t)(64 + jt)  * KT_ALL * 256 + lane * 8;
                const unsigned* brN = Wp + (size_t)(128 + jt) * KT_ALL * 256 + lane * 8;

                // ---- K over h (kt 0..31): A from LDS ----
                for (int kt = 0; kt < KT_H; ++kt) {
                    Frag a, bR, bZ, bN;
                    const u32x4* ap = (const u32x4*)(arow + kt * 16 + sel4);
                    a.q[0] = ap[0]; a.q[1] = ap[2];
                    const u32x4* p;
                    p = (const u32x4*)(brR + kt * 256); bR.q[0] = p[0]; bR.q[1] = p[1];
                    p = (const u32x4*)(brZ + kt * 256); bZ.q[0] = p[0]; bZ.q[1] = p[1];
                    p = (const u32x4*)(brN + kt * 256); bN.q[0] = p[0]; bN.q[1] = p[1];
                    __builtin_prefetch(brR + (kt + 1) * 256, 0, 1);
                    accR  = wmma_bf16(a, bR, accR);
                    accZ  = wmma_bf16(a, bZ, accZ);
                    accNH = wmma_bf16(a, bN, accNH);
                }
                // ---- K over x (kt 32..47): A from global bf16 x[t] ----
                const unsigned* xrow =
                    (const unsigned*)xbf + ((size_t)t * BB + m) * (II / 2);
                for (int kt = KT_H; kt < KT_ALL; ++kt) {
                    Frag a, bR, bZ, bN;
                    const u32x4* ap = (const u32x4*)(xrow + (kt - KT_H) * 16 + sel4);
                    a.q[0] = ap[0]; a.q[1] = ap[2];
                    const u32x4* p;
                    p = (const u32x4*)(brR + kt * 256); bR.q[0] = p[0]; bR.q[1] = p[1];
                    p = (const u32x4*)(brZ + kt * 256); bZ.q[0] = p[0]; bZ.q[1] = p[1];
                    p = (const u32x4*)(brN + kt * 256); bN.q[0] = p[0]; bN.q[1] = p[1];
                    accR  = wmma_bf16(a, bR, accR);
                    accZ  = wmma_bf16(a, bZ, accZ);
                    accNX = wmma_bf16(a, bN, accNX);
                }

                // ---- elementwise GRU update + fused logit reduce ----
                const int j = jt * 16 + lh;
                const float brv = bi[j] + bh[j];
                const float bzv = bi[HH + j] + bh[HH + j];
                const float bnx = bi[2 * HH + j];
                const float bnh = bh[2 * HH + j];
                const float wj  = fc_w[dir * HH + j];
                float part[8];
                for (int p2 = 0; p2 < 8; ++p2) {
                    const int brow = mt * 16 + rowoff + p2;
                    float hprev = bf2f(hcur16[brow * HH + j]);
                    float r = sigm(accR[p2] + brv);
                    float z = sigm(accZ[p2] + bzv);
                    float n = tanhf(accNX[p2] + bnx + r * (accNH[p2] + bnh));
                    float hn = (1.0f - z) * n + z * hprev;
                    hnxt16[brow * HH + j] = f2bf(hn);
                    part[p2] = hn * wj;
                }
                // reduce over the 16 lanes sharing the same row group
                for (int s = 1; s < 16; s <<= 1)
                    for (int p2 = 0; p2 < 8; ++p2)
                        part[p2] += __shfl_xor(part[p2], s, 32);
                if (lh == 0) {
                    float* lg = logit + ((size_t)dir * TT + t) * BB;
                    for (int p2 = 0; p2 < 8; ++p2)
                        atomicAdd(&lg[mt * 16 + rowoff + p2], part[p2]);
                }
            }
        }
        __syncthreads();   // h_next visible to all waves before next step
    }
}

// ---------------------------------------------------------------------------
// att[t,b] = sigmoid(SIG_W * (logit_f + logit_b + fc_b)); also writes the
// transposed attention output [B,T] into d_out.
// ---------------------------------------------------------------------------
__global__ void att_combine_kernel(const float* __restrict__ logit,
                                   const float* __restrict__ fc_b,
                                   float* __restrict__ att,
                                   float* __restrict__ out_att) {
    int idx = blockIdx.x * blockDim.x + threadIdx.x;   // T*B, t-major
    if (idx >= TT * BB) return;
    int b = idx % BB, t = idx / BB;
    float v = sigm(SIG_W * (logit[idx] + logit[TT * BB + idx] + fc_b[0]));
    att[idx] = v;
    out_att[(size_t)b * TT + t] = v;
}

// ---------------------------------------------------------------------------
// TAGM attention-gated GRU recurrence (forward only). Same persistent-block
// structure: per (jt, mt): r (Wzr nt=jt), z (Wzr nt=64+jt), candidate
// h-part/x-part (Wt nt=jt).
// ---------------------------------------------------------------------------
__global__ __launch_bounds__(1024) void tagm_kernel(
    const unsigned* __restrict__ wzr_pack, const unsigned* __restrict__ wt_pack,
    const float* __restrict__ bi_zr, const float* __restrict__ bh_zr,
    const float* __restrict__ bi_t,  const float* __restrict__ bh_t,
    const unsigned short* __restrict__ xbf, const float* __restrict__ att,
    float* __restrict__ hlast) {
    extern __shared__ unsigned smem[];
    const int tid  = threadIdx.x;
    const int wave = tid >> 5;
    const int lane = tid & 31;
    const int lh   = lane & 15;
    const int sel4 = (lane & 16) ? 4 : 0;
    const int rowoff = (lane & 16) ? 8 : 0;
    const int HBUF = BB * (HH / 2);

    for (int i = tid; i < HBUF; i += 1024) smem[i] = 0u;
    __syncthreads();

    for (int t = 0; t < TT; ++t) {
        const unsigned* hcur = smem + (t & 1) * HBUF;
        unsigned*       hnxt = smem + ((t & 1) ^ 1) * HBUF;
        const unsigned short* hcur16 = (const unsigned short*)hcur;
        unsigned short*       hnxt16 = (unsigned short*)hnxt;

        for (int jt2 = 0; jt2 < 2; ++jt2) {
            const int jt = 2 * wave + jt2;
            for (int mt = 0; mt < 4; ++mt) {
                const int m = mt * 16 + lh;
                f32x8 accR = zero8(), accZ = zero8();
                f32x8 accTH = zero8(), accTX = zero8();

                const unsigned* arow = hcur + m * (HH / 2);
                const unsigned* brR = wzr_pack + (size_t)(jt)      * KT_ALL * 256 + lane * 8;
                const unsigned* brZ = wzr_pack + (size_t)(64 + jt) * KT_ALL * 256 + lane * 8;
                const unsigned* brT = wt_pack  + (size_t)(jt)      * KT_ALL * 256 + lane * 8;

                for (int kt = 0; kt < KT_H; ++kt) {
                    Frag a, bR, bZ, bT;
                    const u32x4* ap = (const u32x4*)(arow + kt * 16 + sel4);
                    a.q[0] = ap[0]; a.q[1] = ap[2];
                    const u32x4* p;
                    p = (const u32x4*)(brR + kt * 256); bR.q[0] = p[0]; bR.q[1] = p[1];
                    p = (const u32x4*)(brZ + kt * 256); bZ.q[0] = p[0]; bZ.q[1] = p[1];
                    p = (const u32x4*)(brT + kt * 256); bT.q[0] = p[0]; bT.q[1] = p[1];
                    __builtin_prefetch(brR + (kt + 1) * 256, 0, 1);
                    accR  = wmma_bf16(a, bR, accR);
                    accZ  = wmma_bf16(a, bZ, accZ);
                    accTH = wmma_bf16(a, bT, accTH);
                }
                const unsigned* xrow =
                    (const unsigned*)xbf + ((size_t)t * BB + m) * (II / 2);
                for (int kt = KT_H; kt < KT_ALL; ++kt) {
                    Frag a, bR, bZ, bT;
                    const u32x4* ap = (const u32x4*)(xrow + (kt - KT_H) * 16 + sel4);
                    a.q[0] = ap[0]; a.q[1] = ap[2];
                    const u32x4* p;
                    p = (const u32x4*)(brR + kt * 256); bR.q[0] = p[0]; bR.q[1] = p[1];
                    p = (const u32x4*)(brZ + kt * 256); bZ.q[0] = p[0]; bZ.q[1] = p[1];
                    p = (const u32x4*)(brT + kt * 256); bT.q[0] = p[0]; bT.q[1] = p[1];
                    accR  = wmma_bf16(a, bR, accR);
                    accZ  = wmma_bf16(a, bZ, accZ);
                    accTX = wmma_bf16(a, bT, accTX);
                }

                const int j = jt * 16 + lh;
                const float brv = bi_zr[j] + bh_zr[j];
                const float bzv = bi_zr[HH + j] + bh_zr[HH + j];
                const float btx = bi_t[j];
                const float bth = bh_t[j];
                for (int p2 = 0; p2 < 8; ++p2) {
                    const int brow = mt * 16 + rowoff + p2;
                    float hprev = bf2f(hcur16[brow * HH + j]);
                    float r  = sigm(accR[p2] + brv);
                    float z  = sigm(accZ[p2] + bzv);
                    float ht = tanhf(accTX[p2] + btx + r * (accTH[p2] + bth));
                    float a  = att[(size_t)t * BB + brow];
                    float hn = a * (z * ht + (1.0f - z) * hprev);
                    hnxt16[brow * HH + j] = f2bf(hn);
                }
            }
        }
        __syncthreads();
    }
    // T = 512 (even): final h lives in buffer 0
    const unsigned short* hf = (const unsigned short*)smem;
    for (int i = tid; i < BB * HH; i += 1024) hlast[i] = bf2f(hf[i]);
}

// ---------------------------------------------------------------------------
// out[b,c] = h_last[b,:] . fc0_w[c,:] + fc0_b[c]   (64 x 128 x 1024 — tiny)
// ---------------------------------------------------------------------------
__global__ void fc_kernel(const float* __restrict__ hlast,
                          const float* __restrict__ w,
                          const float* __restrict__ bias,
                          float* __restrict__ out) {
    int b = blockIdx.x, c = threadIdx.x;
    const float* hr = hlast + (size_t)b * HH;
    const float* wr = w + (size_t)c * HH;
    float acc = bias[c];
    for (int j = 0; j < HH; ++j) acc = fmaf(hr[j], wr[j], acc);
    out[b * CC + c] = acc;
}

// ---------------------------------------------------------------------------
extern "C" void kernel_launch(void* const* d_in, const int* in_sizes, int n_in,
                              void* d_out, int out_size, void* d_ws, size_t ws_size,
                              hipStream_t stream) {
    const float* x        = (const float*)d_in[0];
    const float* att_wi_f = (const float*)d_in[1];
    const float* att_wh_f = (const float*)d_in[2];
    const float* att_bi_f = (const float*)d_in[3];
    const float* att_bh_f = (const float*)d_in[4];
    const float* att_wi_b = (const float*)d_in[5];
    const float* att_wh_b = (const float*)d_in[6];
    const float* att_bi_b = (const float*)d_in[7];
    const float* att_bh_b = (const float*)d_in[8];
    const float* att_fc_w = (const float*)d_in[9];
    const float* att_fc_b = (const float*)d_in[10];
    const float* w_i2h_zr = (const float*)d_in[11];
    const float* b_i2h_zr = (const float*)d_in[12];
    const float* w_h2h_zr = (const float*)d_in[13];
    const float* b_h2h_zr = (const float*)d_in[14];
    const float* w_i2h_t  = (const float*)d_in[15];
    const float* b_i2h_t  = (const float*)d_in[16];
    const float* w_h2h_t  = (const float*)d_in[17];
    const float* b_h2h_t  = (const float*)d_in[18];
    const float* fc0_w    = (const float*)d_in[19];
    const float* fc0_b    = (const float*)d_in[20];

    // ---- workspace layout (bytes) ----
    char* ws = (char*)d_ws;
    unsigned short* xbf     = (unsigned short*)(ws);                    // 33,554,432 B
    unsigned*       wf_pack = (unsigned*)(ws + 33554432);               //  9,437,184 B
    unsigned*       wb_pack = (unsigned*)(ws + 42991616);               //  9,437,184 B
    unsigned*       wzr_pack= (unsigned*)(ws + 52428800);               //  6,291,456 B
    unsigned*       wt_pack = (unsigned*)(ws + 58720256);               //  3,145,728 B
    float*          logit   = (float*)(ws + 61865984);                  //    262,144 B
    float*          att     = (float*)(ws + 62128128);                  //    131,072 B
    float*          hlast   = (float*)(ws + 62259200);                  //    262,144 B

    float* out     = (float*)d_out;          // [B,C] = 8192 floats
    float* out_att = out + BB * CC;          // [B,T] = 32768 floats

    // 1) convert x -> bf16, t-major
    pack_x_kernel<<<(TT * BB * II) / 256, 256, 0, stream>>>(x, xbf);

    // 2) pack weights into WMMA B-fragment layout (K = [h | x])
    pack_w_kernel<<<192 * KT_ALL, 256, 0, stream>>>(att_wh_f, att_wi_f, wf_pack, 192);
    pack_w_kernel<<<192 * KT_ALL, 256, 0, stream>>>(att_wh_b, att_wi_b, wb_pack, 192);
    pack_w_kernel<<<128 * KT_ALL, 256, 0, stream>>>(w_h2h_zr, w_i2h_zr, wzr_pack, 128);
    pack_w_kernel<<< 64 * KT_ALL, 256, 0, stream>>>(w_h2h_t,  w_i2h_t,  wt_pack,  64);

    // 3) zero logit accumulators (atomicAdd targets; deterministic per call)
    zero_kernel<<<(2 * TT * BB + 255) / 256, 256, 0, stream>>>(logit, 2 * TT * BB);

    // 4) bidirectional attention GRU (persistent, 256 KB LDS ping-pong h)
    att_gru_kernel<<<2, 1024, 2 * BB * HH * 2, stream>>>(
        wf_pack, wb_pack, att_bi_f, att_bh_f, att_bi_b, att_bh_b,
        xbf, att_fc_w, logit);

    // 5) attention combine (+ transposed att output)
    att_combine_kernel<<<(TT * BB + 255) / 256, 256, 0, stream>>>(
        logit, att_fc_b, att, out_att);

    // 6) TAGM gated recurrence
    tagm_kernel<<<1, 1024, 2 * BB * HH * 2, stream>>>(
        wzr_pack, wt_pack, b_i2h_zr, b_h2h_zr, b_i2h_t, b_h2h_t,
        xbf, att, hlast);

    // 7) final FC
    fc_kernel<<<BB, CC, 0, stream>>>(hlast, fc0_w, fc0_b, out);
}
